// LandmarkAutoencoder_12017318494754
// MI455X (gfx1250) — matrix operand order repaired and checked by reading, not compiled, not run on hardware
//
#include <hip/hip_runtime.h>

// ---------------------------------------------------------------------------
// LandmarkAutoencoder forward for MI455X (gfx1250, wave32, WMMA bf16)
// Tiled WMMA GEMM: M-tile=128, wave grid (8/MT) x (NT/NTW), each wave owns
// MT x NTW 16x16 accumulator tiles. K streamed in 32-slices through
// double-buffered LDS; next slice staged with GLOBAL_LOAD_ASYNC_TO_LDS
// (ASYNCcnt) when available, overlapping the WMMA burst; all A/B fragments
// of a slice loaded up-front so ds waits amortize over the 16-WMMA burst.
// ---------------------------------------------------------------------------

typedef __attribute__((ext_vector_type(16))) __bf16 v16bf;
typedef __attribute__((ext_vector_type(8)))  float  v8f;
typedef __attribute__((ext_vector_type(4)))  unsigned u32x4;
typedef __attribute__((ext_vector_type(4)))  int      i32x4;

#if defined(__AMDGCN__) && __has_builtin(__builtin_amdgcn_global_load_async_to_lds_b128)
#define HAVE_ASYNC_LDS 1
typedef __attribute__((address_space(1))) i32x4* gv4i_ptr;  // global v4i*
typedef __attribute__((address_space(3))) i32x4* lv4i_ptr;  // LDS v4i*
static __device__ __forceinline__ void async_cp16(const void* g, void* l) {
  __builtin_amdgcn_global_load_async_to_lds_b128(
      (gv4i_ptr)(g), (lv4i_ptr)(l), 0, 0);
}
static __device__ __forceinline__ void wait_async0() {
#if __has_builtin(__builtin_amdgcn_s_wait_asynccnt)
  __builtin_amdgcn_s_wait_asynccnt(0);
#else
  asm volatile("s_wait_asynccnt 0" ::: "memory");
#endif
}
#else
#define HAVE_ASYNC_LDS 0
#endif

static __device__ __forceinline__ __bf16 f2bf(float f) {
  unsigned u = __builtin_bit_cast(unsigned, f);
  u += 0x7FFFu + ((u >> 16) & 1u);          // round-to-nearest-even
  unsigned short h = (unsigned short)(u >> 16);
  return __builtin_bit_cast(__bf16, h);
}

union V16U { u32x4 q[2]; v16bf v; };

// ---------------------------------------------------------------------------
// GEMM: O[M,N] = act(A[M,K] @ W^T + bias).  W stored transposed: [N][K] bf16.
// K multiple of 32. N = NT*16 (padded); stores guarded by ncols.
// ---------------------------------------------------------------------------
template<int NT, int MT, int NTW, bool RELU, bool F32OUT>
__global__ __launch_bounds__(256)
void gemm_bias_act(const __bf16* __restrict__ A, int lda,
                   const __bf16* __restrict__ W, const float* __restrict__ bias,
                   int K, void* __restrict__ OutP, int ldo, int ncols)
{
  constexpr int N = NT * 16;
  static_assert((8 % MT) == 0 && (NT % NTW) == 0, "bad wave tiling");
  static_assert((8 / MT) * (NT / NTW) == 8, "wave grid must be 8 waves");
  constexpr int WROWS = 8 / MT;             // waves along M

  __shared__ __bf16 sA[2][128 * 32];
  __shared__ __bf16 sB[2][N * 32];

  const int t    = threadIdx.x;
  const int wave = t >> 5;
  const int lane = t & 31;
  const int l15  = lane & 15;
  const int hi   = lane >> 4;
  const int m0   = blockIdx.x * 128;

  const int mtile0 = (wave % WROWS) * MT;   // first M 16-tile of this wave
  const int ntile0 = (wave / WROWS) * NTW;  // first N 16-tile of this wave

  v8f acc[MT][NTW];
#pragma unroll
  for (int i = 0; i < MT; ++i)
#pragma unroll
    for (int j = 0; j < NTW; ++j)
#pragma unroll
      for (int r = 0; r < 8; ++r) acc[i][j][r] = 0.0f;

  const int aRow   = t >> 1;                // 0..127
  const int aHalf  = (t & 1) * 16;          // 0 or 16
  const int aBase1 = hi ? 8 : 0;            // A frag K chunks {b..b+7, b+16..}
  const int bKoff  = hi ? 8 : 0;            // B frag K chunks {b..b+7, b+16..}

  const __bf16* aGlob = A + (long)(m0 + aRow) * lda + aHalf;
  const __bf16* bGlob = W + (long)t * K;

  // ---- stage slice 0 into buffer 0 ----
#if HAVE_ASYNC_LDS
  {
    __bf16* adst = &sA[0][aRow * 32 + aHalf];
    async_cp16(aGlob,     adst);
    async_cp16(aGlob + 8, adst + 8);
    if (t < N) {
      __bf16* bdst = &sB[0][t * 32];
      async_cp16(bGlob,      bdst);
      async_cp16(bGlob + 8,  bdst + 8);
      async_cp16(bGlob + 16, bdst + 16);
      async_cp16(bGlob + 24, bdst + 24);
    }
    wait_async0();
  }
#else
  {
    const u32x4* asrc = (const u32x4*)(aGlob);
    u32x4* adst = (u32x4*)(&sA[0][aRow * 32 + aHalf]);
    adst[0] = asrc[0];
    adst[1] = asrc[1];
    if (t < N) {
      const u32x4* bsrc = (const u32x4*)(bGlob);
      u32x4* bdst = (u32x4*)(&sB[0][t * 32]);
      bdst[0] = bsrc[0]; bdst[1] = bsrc[1]; bdst[2] = bsrc[2]; bdst[3] = bsrc[3];
    }
  }
#endif
  __syncthreads();

  const int S = K >> 5;                     // number of 32-wide K slices
  for (int s = 0; s < S; ++s) {
    const int cur = s & 1;
    const int nxt = cur ^ 1;
    const bool haveNext = (s + 1) < S;

    // ---- stage next slice early so the copy overlaps the WMMA burst ----
#if HAVE_ASYNC_LDS
    if (haveNext) {
      const int kb = (s + 1) << 5;
      __bf16* adst = &sA[nxt][aRow * 32 + aHalf];
      async_cp16(aGlob + kb,     adst);
      async_cp16(aGlob + kb + 8, adst + 8);
      if (t < N) {
        __bf16* bdst = &sB[nxt][t * 32];
        async_cp16(bGlob + kb,      bdst);
        async_cp16(bGlob + kb + 8,  bdst + 8);
        async_cp16(bGlob + kb + 16, bdst + 16);
        async_cp16(bGlob + kb + 24, bdst + 24);
      }
    }
#else
    u32x4 ga0, ga1, gb0, gb1, gb2, gb3;
    if (haveNext) {
      const int kb = (s + 1) << 5;
      const u32x4* asrc = (const u32x4*)(aGlob + kb);
      ga0 = asrc[0]; ga1 = asrc[1];
      if (t < N) {
        const u32x4* bsrc = (const u32x4*)(bGlob + kb);
        gb0 = bsrc[0]; gb1 = bsrc[1]; gb2 = bsrc[2]; gb3 = bsrc[3];
      }
    }
#endif

    // ---- compute on current buffer: all fragments up-front, then burst ----
    {
      const __bf16* sAc = sA[cur];
      const __bf16* sBc = sB[cur];

      V16U af[MT];
#pragma unroll
      for (int mt = 0; mt < MT; ++mt) {
        const __bf16* ap = sAc + ((mtile0 + mt) * 16 + l15) * 32;
        af[mt].q[0] = *(const u32x4*)(ap + aBase1);
        af[mt].q[1] = *(const u32x4*)(ap + aBase1 + 16);
      }
      V16U bf[NTW];
#pragma unroll
      for (int j = 0; j < NTW; ++j) {
        const __bf16* bp = sBc + ((ntile0 + j) * 16 + l15) * 32 + bKoff;
        bf[j].q[0] = *(const u32x4*)(bp);
        bf[j].q[1] = *(const u32x4*)(bp + 16);
      }
#pragma unroll
      for (int j = 0; j < NTW; ++j) {
#pragma unroll
        for (int mt = 0; mt < MT; ++mt) {
          acc[mt][j] = __builtin_amdgcn_wmma_f32_16x16x32_bf16(
              false, af[mt].v, false, bf[j].v, (short)0, acc[mt][j], false, false);
        }
      }
    }

    // ---- make next buffer visible, then sync ----
    if (haveNext) {
#if HAVE_ASYNC_LDS
      wait_async0();
#else
      u32x4* adst = (u32x4*)(&sA[nxt][aRow * 32 + aHalf]);
      adst[0] = ga0;
      adst[1] = ga1;
      if (t < N) {
        u32x4* bdst = (u32x4*)(&sB[nxt][t * 32]);
        bdst[0] = gb0; bdst[1] = gb1; bdst[2] = gb2; bdst[3] = gb3;
      }
#endif
      __syncthreads();
    }
  }

  // ---- epilogue: C layout: VGPR r -> row r (lanes 0-15) / 8+r (lanes 16-31)
#pragma unroll
  for (int mt = 0; mt < MT; ++mt) {
#pragma unroll
    for (int j = 0; j < NTW; ++j) {
      const int col = (ntile0 + j) * 16 + l15;
      if (col < ncols) {
        const float bv = bias[col];
#pragma unroll
        for (int r = 0; r < 8; ++r) {
          const long row = m0 + (mtile0 + mt) * 16 + (hi ? 8 + r : r);
          float v = acc[mt][j][r] + bv;
          if (RELU) v = v > 0.0f ? v : 0.0f;
          if (F32OUT) ((float*)OutP)[row * (long)ldo + col] = v;
          else        ((__bf16*)OutP)[row * (long)ldo + col] = f2bf(v);
        }
      }
    }
  }
}

// fp32 [B,136] -> bf16 [B,160] zero-padded
__global__ void prep_input_kernel(const float* __restrict__ in,
                                  __bf16* __restrict__ out, long B)
{
  long i = (long)blockIdx.x * blockDim.x + threadIdx.x;
  if (i >= B * 160) return;
  long row = i / 160; int c = (int)(i % 160);
  out[i] = (c < 136) ? f2bf(in[row * 136 + c]) : f2bf(0.0f);
}

// W [K][N] fp32 (row-major, x@W) -> WT [Npad][Kpad] bf16 (transposed, padded)
__global__ void prep_weight_kernel(const float* __restrict__ W,
                                   const float* __restrict__ b,
                                   int K, int N,
                                   __bf16* __restrict__ WT,
                                   float* __restrict__ bout,
                                   int Kpad, int Npad)
{
  int i = blockIdx.x * blockDim.x + threadIdx.x;
  if (i >= Npad * Kpad) return;
  int n = i / Kpad, k = i % Kpad;
  WT[i] = (n < N && k < K) ? f2bf(W[(long)k * N + n]) : f2bf(0.0f);
  if (k == 0) bout[n] = (n < N) ? b[n] : 0.0f;
}

// In-place mouth-embedding swap on E = [eye|mouth] [B,128] for streams a,b
__global__ void mix_kernel(__bf16* __restrict__ Ea, __bf16* __restrict__ Eb,
                           const int* __restrict__ sw, long B)
{
  long i = (long)blockIdx.x * blockDim.x + threadIdx.x;
  if (i >= B * 64) return;
  long row = i / 64; int c = 64 + (int)(i % 64);
  long idx = row * 128 + c;
  __bf16 ma = Ea[idx], mb = Eb[idx];
  if (sw[row] == 1) { Ea[idx] = mb; Eb[idx] = ma; }
}

__global__ void zero3_kernel(float* p) { if (threadIdx.x < 3) p[threadIdx.x] = 0.0f; }

// Mouth-landmark swap + L1 loss reduction
__global__ __launch_bounds__(256)
void loss_kernel(const float* __restrict__ la, const float* __restrict__ lb,
                 const int* __restrict__ sw,
                 const float* __restrict__ pa, const float* __restrict__ pb,
                 float* __restrict__ laN, float* __restrict__ lbN,
                 float* __restrict__ loss, long B, float invB)
{
  __shared__ float ra[256];
  __shared__ float rb[256];
  const int t = threadIdx.x;
  long i = (long)blockIdx.x * 256 + t;
  float s_a = 0.0f, s_b = 0.0f;
  if (i < B * 68) {
    long row = i / 68; int pt = (int)(i % 68);
    long o = i * 2;
    float ax = la[o], ay = la[o + 1], bx = lb[o], by = lb[o + 1];
    bool m = (sw[row] == 1) && (pt >= 48);
    float nax = m ? bx : ax, nay = m ? by : ay;
    float nbx = m ? ax : bx, nby = m ? ay : by;
    laN[o] = nax; laN[o + 1] = nay;
    lbN[o] = nbx; lbN[o + 1] = nby;
    s_a = fabsf(nax - pa[o]) + fabsf(nay - pa[o + 1]);
    s_b = fabsf(nbx - pb[o]) + fabsf(nby - pb[o + 1]);
  }
  ra[t] = s_a; rb[t] = s_b;
  __syncthreads();
  for (int s = 128; s > 0; s >>= 1) {
    if (t < s) { ra[t] += ra[t + s]; rb[t] += rb[t + s]; }
    __syncthreads();
  }
  if (t == 0) {
    atomicAdd(&loss[0], ra[0] * invB);
    atomicAdd(&loss[1], rb[0] * invB);
  }
}

__global__ void finalize_kernel(float* loss) { loss[2] = loss[0] + loss[1]; }

// ---------------------------------------------------------------------------
// Host driver
// ---------------------------------------------------------------------------
extern "C" void kernel_launch(void* const* d_in, const int* in_sizes, int n_in,
                              void* d_out, int out_size, void* d_ws, size_t ws_size,
                              hipStream_t stream)
{
  (void)n_in; (void)out_size; (void)ws_size;
  const long B = (long)in_sizes[0] / 136;           // 65536
  const int  GB = (int)(B / 128);                   // GEMM grid (M-tiles)

  // Inputs in setup_inputs() dict insertion order
  const float* lmA = (const float*)d_in[0];
  const float* lmB = (const float*)d_in[1];
  const int*   sw  = (const int*)d_in[2];
  const float *ew[5], *eb[5];
  for (int i = 0; i < 5; ++i) { ew[i] = (const float*)d_in[3 + 2 * i]; eb[i] = (const float*)d_in[4 + 2 * i]; }
  const float* we = (const float*)d_in[13]; const float* be = (const float*)d_in[14];
  const float* wm = (const float*)d_in[15]; const float* bm = (const float*)d_in[16];
  const float *dw[6], *db[6];
  for (int j = 0; j < 6; ++j) { dw[j] = (const float*)d_in[17 + 2 * j]; db[j] = (const float*)d_in[18 + 2 * j]; }

  // Workspace carve-up
  char* ws = (char*)d_ws;
  size_t cur = 0;
  auto alloc = [&](size_t bytes) -> char* {
    char* p = ws + cur;
    cur = (cur + bytes + 255) & ~(size_t)255;
    return p;
  };

  // Encoder layers: K x N (padded Kp x Np)
  const int eK[5]  = {136, 256, 256, 256, 256};
  const int eN[5]  = {256, 256, 256, 256, 128};
  const int eKp[5] = {160, 256, 256, 256, 256};
  __bf16* wtE[5]; float* bE[5];
  for (int i = 0; i < 5; ++i) {
    wtE[i] = (__bf16*)alloc((size_t)eN[i] * eKp[i] * 2);
    bE[i]  = (float*)alloc((size_t)eN[i] * 4);
  }
  __bf16* wtH = (__bf16*)alloc(128 * 128 * 2);      // [we|wm] fused head
  float*  bH  = (float*)alloc(128 * 4);

  const int dK[6]  = {128, 256, 256, 256, 256, 256};
  const int dN[6]  = {256, 256, 256, 256, 256, 136};
  const int dNp[6] = {256, 256, 256, 256, 256, 144};
  __bf16* wtD[6]; float* bD[6];
  for (int j = 0; j < 6; ++j) {
    wtD[j] = (__bf16*)alloc((size_t)dNp[j] * dK[j] * 2);
    bD[j]  = (float*)alloc((size_t)dNp[j] * 4);
  }

  __bf16* x0   = (__bf16*)alloc((size_t)B * 160 * 2);
  __bf16* buf0 = (__bf16*)alloc((size_t)B * 256 * 2);
  __bf16* buf1 = (__bf16*)alloc((size_t)B * 256 * 2);
  __bf16* Ea   = (__bf16*)alloc((size_t)B * 128 * 2);
  __bf16* Eb   = (__bf16*)alloc((size_t)B * 128 * 2);

  // --- weight prep (transpose + bf16 + pad) ---
  for (int i = 0; i < 5; ++i) {
    int n = eN[i] * eKp[i];
    prep_weight_kernel<<<(n + 255) / 256, 256, 0, stream>>>(
        ew[i], eb[i], eK[i], eN[i], wtE[i], bE[i], eKp[i], eN[i]);
  }
  prep_weight_kernel<<<(64 * 128 + 255) / 256, 256, 0, stream>>>(
      we, be, 128, 64, wtH, bH, 128, 64);
  prep_weight_kernel<<<(64 * 128 + 255) / 256, 256, 0, stream>>>(
      wm, bm, 128, 64, wtH + 64 * 128, bH + 64, 128, 64);
  for (int j = 0; j < 6; ++j) {
    int n = dNp[j] * dK[j];
    prep_weight_kernel<<<(n + 255) / 256, 256, 0, stream>>>(
        dw[j], db[j], dK[j], dN[j], wtD[j], bD[j], dK[j], dNp[j]);
  }

  // --- output layout ---
  float* out  = (float*)d_out;
  const long BP = B * 136;
  float* loss = out;
  float* pa   = out + 3;
  float* pb   = pa + BP;
  float* laN  = pb + BP;
  float* lbN  = laN + BP;

  const long nPrep = B * 160;

  // --- encoder, stream a ---
  prep_input_kernel<<<(int)((nPrep + 255) / 256), 256, 0, stream>>>(lmA, x0, B);
  gemm_bias_act<16, 4, 4, true,  false><<<GB, 256, 0, stream>>>(x0,   160, wtE[0], bE[0], 160, buf0, 256, 256);
  gemm_bias_act<16, 4, 4, true,  false><<<GB, 256, 0, stream>>>(buf0, 256, wtE[1], bE[1], 256, buf1, 256, 256);
  gemm_bias_act<16, 4, 4, true,  false><<<GB, 256, 0, stream>>>(buf1, 256, wtE[2], bE[2], 256, buf0, 256, 256);
  gemm_bias_act<16, 4, 4, true,  false><<<GB, 256, 0, stream>>>(buf0, 256, wtE[3], bE[3], 256, buf1, 256, 256);
  gemm_bias_act<8,  4, 2, true,  false><<<GB, 256, 0, stream>>>(buf1, 256, wtE[4], bE[4], 256, buf0, 128, 128);
  gemm_bias_act<8,  4, 2, false, false><<<GB, 256, 0, stream>>>(buf0, 128, wtH,    bH,    128, Ea,   128, 128);

  // --- encoder, stream b ---
  prep_input_kernel<<<(int)((nPrep + 255) / 256), 256, 0, stream>>>(lmB, x0, B);
  gemm_bias_act<16, 4, 4, true,  false><<<GB, 256, 0, stream>>>(x0,   160, wtE[0], bE[0], 160, buf0, 256, 256);
  gemm_bias_act<16, 4, 4, true,  false><<<GB, 256, 0, stream>>>(buf0, 256, wtE[1], bE[1], 256, buf1, 256, 256);
  gemm_bias_act<16, 4, 4, true,  false><<<GB, 256, 0, stream>>>(buf1, 256, wtE[2], bE[2], 256, buf0, 256, 256);
  gemm_bias_act<16, 4, 4, true,  false><<<GB, 256, 0, stream>>>(buf0, 256, wtE[3], bE[3], 256, buf1, 256, 256);
  gemm_bias_act<8,  4, 2, true,  false><<<GB, 256, 0, stream>>>(buf1, 256, wtE[4], bE[4], 256, buf0, 128, 128);
  gemm_bias_act<8,  4, 2, false, false><<<GB, 256, 0, stream>>>(buf0, 128, wtH,    bH,    128, Eb,   128, 128);

  // --- switch mix (in-place on mouth halves) ---
  mix_kernel<<<(int)((B * 64 + 255) / 256), 256, 0, stream>>>(Ea, Eb, sw, B);

  // --- decoder, stream a -> pred_a (f32 into d_out) ---
  gemm_bias_act<16, 4, 4, true,  false><<<GB, 256, 0, stream>>>(Ea,   128, wtD[0], bD[0], 128, buf0, 256, 256);
  gemm_bias_act<16, 4, 4, true,  false><<<GB, 256, 0, stream>>>(buf0, 256, wtD[1], bD[1], 256, buf1, 256, 256);
  gemm_bias_act<16, 4, 4, true,  false><<<GB, 256, 0, stream>>>(buf1, 256, wtD[2], bD[2], 256, buf0, 256, 256);
  gemm_bias_act<16, 4, 4, true,  false><<<GB, 256, 0, stream>>>(buf0, 256, wtD[3], bD[3], 256, buf1, 256, 256);
  gemm_bias_act<16, 4, 4, true,  false><<<GB, 256, 0, stream>>>(buf1, 256, wtD[4], bD[4], 256, buf0, 256, 256);
  gemm_bias_act<9,  1, 9, false, true ><<<GB, 256, 0, stream>>>(buf0, 256, wtD[5], bD[5], 256, pa,   136, 136);

  // --- decoder, stream b -> pred_b ---
  gemm_bias_act<16, 4, 4, true,  false><<<GB, 256, 0, stream>>>(Eb,   128, wtD[0], bD[0], 128, buf0, 256, 256);
  gemm_bias_act<16, 4, 4, true,  false><<<GB, 256, 0, stream>>>(buf0, 256, wtD[1], bD[1], 256, buf1, 256, 256);
  gemm_bias_act<16, 4, 4, true,  false><<<GB, 256, 0, stream>>>(buf1, 256, wtD[2], bD[2], 256, buf0, 256, 256);
  gemm_bias_act<16, 4, 4, true,  false><<<GB, 256, 0, stream>>>(buf0, 256, wtD[3], bD[3], 256, buf1, 256, 256);
  gemm_bias_act<16, 4, 4, true,  false><<<GB, 256, 0, stream>>>(buf1, 256, wtD[4], bD[4], 256, buf0, 256, 256);
  gemm_bias_act<9,  1, 9, false, true ><<<GB, 256, 0, stream>>>(buf0, 256, wtD[5], bD[5], 256, pb,   136, 136);

  // --- landmark swap + L1 losses ---
  zero3_kernel<<<1, 32, 0, stream>>>(loss);
  loss_kernel<<<(int)((B * 68 + 255) / 256), 256, 0, stream>>>(
      lmA, lmB, sw, pa, pb, laN, lbN, loss, B, 1.0f / (float)B);
  finalize_kernel<<<1, 1, 0, stream>>>(loss);
}
